// HdcGenericEncoder_90692529422509
// MI455X (gfx1250) — matrix operand order repaired and checked by reading, not compiled, not run on hardware
//
#include <hip/hip_runtime.h>

typedef __attribute__((ext_vector_type(16))) _Float16 v16h;
typedef __attribute__((ext_vector_type(8)))  float    v8f;

#define HDC_D     10000
#define HDC_C     32
#define HDC_NG    7
#define HDC_W     25              // C - NGRAMS
#define TILE_D    128
#define HALO      6               // NGRAMS - 1
#define ROW_LEN   (TILE_D + HALO) // 134

__global__ __launch_bounds__(256)
void hdc_encode_wmma(const float* __restrict__ channels,
                     const float* __restrict__ keys,
                     const float* __restrict__ motion_table,
                     const float* __restrict__ hr_table,
                     float* __restrict__ out)
{
    __shared__ const float* rowptr[HDC_C];
    __shared__ unsigned int signw[ROW_LEN];  // bit c = sign(level0[b,c,p]), c=1..31
    __shared__ unsigned int keyw[TILE_D];    // bit w = sign(keys[w,d]), w=0..24 (25..31 = 0)

    const int tid = threadIdx.x;
    const int b   = blockIdx.y;
    const int s   = blockIdx.x * TILE_D;     // start of this block's d-tile

    // ---- level-lookup: quantize channels[b, t=0, c] -> table row pointer ----
    if (tid < HDC_C) {
        const float x = channels[b * (4 * HDC_C) + tid];   // T=0 slice
        const float* p;
        if (tid < HDC_C - 1) {
            int idx = __float2int_rn((x - (-3.0f)) / 6.0f * 2999.0f);  // round-half-even
            idx = idx < 0 ? 0 : (idx > 2999 ? 2999 : idx);
            p = motion_table + (size_t)idx * HDC_D;
        } else {
            int idx = __float2int_rn((x - 50.0f) / 150.0f * 199.0f);
            idx = idx < 0 ? 0 : (idx > 199 ? 199 : idx);
            p = hr_table + (size_t)idx * HDC_D;
        }
        rowptr[tid] = p;
    }
    __syncthreads();

    // ---- pack sign bits of rows c=1..31 over positions [s-6, s+TILE) (mod D) ----
    if (tid < ROW_LEN) {
        int p = s - HALO + tid;
        if (p < 0)      p += HDC_D;
        if (p >= HDC_D) p -= HDC_D;
        unsigned int w = 0;
#pragma unroll
        for (int c = 1; c < HDC_C; ++c)                 // coalesced across threads
            w |= (rowptr[c][p] < 0.0f ? 1u : 0u) << c;
        signw[tid] = w;
    }
    // ---- pack sign bits of keys rows 0..24 over [s, s+TILE) ----
    if (tid < TILE_D) {
        int d = s + tid;
        if (d >= HDC_D) d -= HDC_D;                     // edge tile: wrapped, unused
        unsigned int w = 0;
#pragma unroll
        for (int k = 0; k < HDC_W; ++k)                 // coalesced across threads
            w |= (keys[k * HDC_D + d] < 0.0f ? 1u : 0u) << k;
        keyw[tid] = w;
    }
    __syncthreads();

    // ---- one v_wmma_f32_16x16x32_f16 per wave: M = 16 d-values, K = 25 windows ----
    const int lane  = tid & 31;
    const int wave  = tid >> 5;
    const int r     = lane & 15;
    const int hi    = lane >> 4;
    const int dbase = wave * 16;     // this wave's d-offset within the tile
    const int drel  = dbase + r;     // d - s for this lane's A-row / B-column

    // bit w of X = XOR_j sign(level0[b, 1+w+j, d-6+j])  -> all 25 ngram signs at once
    unsigned int X = 0;
#pragma unroll
    for (int j = 0; j < HDC_NG; ++j)
        X ^= signw[drel + j] >> (1 + j);

    // Hoist the per-lane variable shifts once; per-half extracts use constant shifts.
    // A layout: lane r = row M=r, K(h) = h + 16*hi               -> bit h of XA
    // B layout: lane r = col N=r, K(h) = h + 16*hi               -> bit h of KB
    const unsigned int XA = X >> (16 * hi);
    const unsigned int KB = keyw[drel] >> (16 * hi);

    v16h A;
    v16h B;
#pragma unroll
    for (int h = 0; h < 16; ++h) {
        const int pa = h;
        // K >= 25 rows of A may be junk +/-1: the matching B rows are zeroed below,
        // so those products are exactly 0 in the WMMA.
        A[h] = ((XA >> pa) & 1u) ? (_Float16)-1.0f : (_Float16)1.0f;

        const _Float16 bv = ((KB >> h) & 1u) ? (_Float16)-1.0f : (_Float16)1.0f;
        if (h < 9) {
            B[h] = bv;                                  // h+16*hi <= 24 : always valid
        } else {
            B[h] = hi ? (_Float16)0.0f : bv;            // K = h+16 >= 25 : zero-pad
        }
    }

    v8f Cacc = {};
    Cacc = __builtin_amdgcn_wmma_f32_16x16x32_f16(false, A, false, B,
                                                  (short)0, Cacc, false, false);

    // ---- extract diagonal C[m][m] = ms[b, s+dbase+m] and emit sign ----
    // C/D layout: VGPR g -> lanes 0-15: (M=g, N=lane); lanes 16-31: (M=8+g, N=lane-16)
    int g = -1, m = 0;
    if (lane < 8)        { g = lane;      m = lane;      }
    else if (lane >= 24) { g = lane - 24; m = lane - 16; }
    if (g >= 0) {
        float v = 0.0f;
        switch (g) {
            case 0: v = Cacc[0]; break;
            case 1: v = Cacc[1]; break;
            case 2: v = Cacc[2]; break;
            case 3: v = Cacc[3]; break;
            case 4: v = Cacc[4]; break;
            case 5: v = Cacc[5]; break;
            case 6: v = Cacc[6]; break;
            case 7: v = Cacc[7]; break;
        }
        const int d = s + dbase + m;
        if (d < HDC_D)
            out[(size_t)b * HDC_D + d] = (v > 0.0f) ? 1.0f : -1.0f;
    }
}

extern "C" void kernel_launch(void* const* d_in, const int* in_sizes, int n_in,
                              void* d_out, int out_size, void* d_ws, size_t ws_size,
                              hipStream_t stream) {
    const float* channels = (const float*)d_in[0];
    const float* keys     = (const float*)d_in[1];
    const float* mt       = (const float*)d_in[2];
    const float* ht       = (const float*)d_in[3];
    float* out = (float*)d_out;

    const int B = in_sizes[0] / (4 * HDC_C);       // 64
    dim3 grid((HDC_D + TILE_D - 1) / TILE_D, B);   // 79 x 64
    hdc_encode_wmma<<<grid, dim3(256), 0, stream>>>(channels, keys, mt, ht, out);
}
